// DeepRdRp_attention_16458314678842
// MI455X (gfx1250) — matrix operand order repaired and checked by reading, not compiled, not run on hardware
//
#include <hip/hip_runtime.h>

typedef __attribute__((ext_vector_type(16))) _Float16 v16h;
typedef __attribute__((ext_vector_type(8)))  _Float16 v8h;
typedef __attribute__((ext_vector_type(8)))  float    v8f;
typedef __attribute__((ext_vector_type(4)))  unsigned int v4u;
typedef __attribute__((ext_vector_type(8)))  int      v8i;
typedef __attribute__((ext_vector_type(4)))  int      v4i;

#define B_   32
#define S_   128
#define T_   21
#define H_   128
#define C_   256
#define NFC_ 512
#define NC_  5
#define F2T  42      // 2*T

// ---- workspace layout (bytes) ----
static constexpr size_t OFF_A      = 0;                       // (B,S,H) f32  : 2 MB
static constexpr size_t OFF_B      = OFF_A      + 2097152;    // (B,S,H) f32  : 2 MB
static constexpr size_t OFF_FEATH  = OFF_B      + 2097152;    // (B,S,42) f16 : 344064
static constexpr size_t OFF_CONVWT = OFF_FEATH  + 344064;     // transposed padded conv w : 950272
static constexpr size_t OFF_FC1WH  = OFF_CONVWT + 950272;     // 512*1536 f16 : 1572864
static constexpr size_t OFF_POOL   = OFF_FC1WH  + 1572864;    // (B,1536) f32 : 196608
static constexpr size_t OFF_ZH     = OFF_POOL   + 196608;     // (B,1536) f16 : 98304
static constexpr size_t OFF_Z2     = OFF_ZH     + 98304;      // (B,512)  f32 : 65536

struct Ptr6 { const float* p[6]; };
union AFrag { v16h v; _Float16 h[16]; unsigned u[8]; };

// ---------------- 1) projections a,b = xs . w1^T ----------------
__global__ void proj_kernel(const float* __restrict__ x, const float* __restrict__ w1,
                            float* __restrict__ a_buf, float* __restrict__ b_buf) {
  __shared__ float xr[T_];
  const int h = threadIdx.x;                       // 0..127
  const size_t row = blockIdx.x;                   // b*S + s
  if (h < T_) xr[h] = x[row * T_ + h];
  __syncthreads();
  float sa = 0.f, sb = 0.f;
#pragma unroll
  for (int t = 0; t < T_; ++t) {
    const float xv = xr[t];
    sa += xv * w1[h * F2T + t];
    sb += xv * w1[h * F2T + T_ + t];
  }
  a_buf[row * H_ + h] = sa;
  b_buf[row * H_ + h] = sb;
}

// ------- 2) tanh-score + softmax + att + concat -> feat (f16) -------
__global__ void attn_kernel(const float* __restrict__ x,
                            const float* __restrict__ a_buf, const float* __restrict__ b_buf,
                            const float* __restrict__ b1, const float* __restrict__ w2,
                            const float* __restrict__ b2, _Float16* __restrict__ feat_h) {
  __shared__ float ai[H_], w2s[H_], red[S_], alpha[S_];
  const int j = threadIdx.x;                       // 0..127
  const int b = blockIdx.x / S_;
  const int i = blockIdx.x % S_;
  ai[j]  = a_buf[(size_t)blockIdx.x * H_ + j] + b1[j];
  w2s[j] = w2[j];
  __syncthreads();
  const float* brow = b_buf + ((size_t)(b * S_ + j)) * H_;
  float s = 0.f;
#pragma unroll 4
  for (int h = 0; h < H_; ++h) s += tanhf(ai[h] + brow[h]) * w2s[h];
  s += b2[0];
  if (j == i) s = -1.0e9f;
  red[j] = s; __syncthreads();
  for (int off = 64; off > 0; off >>= 1) { if (j < off) red[j] = fmaxf(red[j], red[j + off]); __syncthreads(); }
  const float m = red[0]; __syncthreads();
  const float e = __expf(s - m);
  red[j] = e; __syncthreads();
  for (int off = 64; off > 0; off >>= 1) { if (j < off) red[j] += red[j + off]; __syncthreads(); }
  alpha[j] = e / red[0];
  __syncthreads();
  if (j < T_) {
    float acc = 0.f;
    for (int jj = 0; jj < S_; ++jj) acc += alpha[jj] * x[((size_t)(b * S_ + jj)) * T_ + j];
    const size_t fo = (size_t)blockIdx.x * F2T;
    feat_h[fo + j]      = (_Float16)x[(size_t)blockIdx.x * T_ + j];
    feat_h[fo + T_ + j] = (_Float16)acc;
  }
}

// ---- 3) transpose+pad conv weights to [n][Kp] f16, fc1 f32->f16, zero pooled ----
__global__ void prep_kernel(Ptr6 cw, const float* __restrict__ fc1w,
                            _Float16* __restrict__ convwT, _Float16* __restrict__ fc1w_h,
                            float* __restrict__ pooled) {
  const int KSarr[6] = {2, 4, 6, 8, 10, 12};
  const int KpArr[6] = {96, 192, 256, 352, 448, 512};
  const int woffT[6] = {0, 24576, 73728, 139264, 229376, 344064};
  const int CONVT = 475136, FCT = 786432;
  const int idx = blockIdx.x * blockDim.x + threadIdx.x;
  if (idx < CONVT) {
    int ki = 0;
#pragma unroll
    for (int q = 1; q < 6; ++q) if (idx >= woffT[q]) ki = q;
    const int idx2 = idx - woffT[ki];
    const int Kp = KpArr[ki];
    const int n = idx2 / Kp, kk = idx2 % Kp;
    const int Kdim = F2T * KSarr[ki];
    convwT[idx] = (kk < Kdim) ? (_Float16)cw.p[ki][(size_t)kk * C_ + n] : (_Float16)0.f;
  } else if (idx < CONVT + FCT) {
    const int jj = idx - CONVT;
    fc1w_h[jj] = (_Float16)fc1w[jj];            // fc1_w (512,1536) row-major == N-major
  } else if (idx < CONVT + FCT + B_ * 1536) {
    pooled[idx - CONVT - FCT] = 0.f;
  }
}

// ---- 4) six convs: TDM-staged A tile + WMMA + ReLU + atomicMax pool ----
__global__ void conv_wmma(const _Float16* __restrict__ feat, const _Float16* __restrict__ convwT,
                          Ptr6 cbias, float* __restrict__ pooled) {
  __shared__ _Float16 ldsA[16 * 504];            // 16 rows x Kdim halfs (packed)
  const int KSarr[6] = {2, 4, 6, 8, 10, 12};
  const int KpArr[6] = {96, 192, 256, 352, 448, 512};
  const int woffT[6] = {0, 24576, 73728, 139264, 229376, 344064};
  const int id = blockIdx.x;                     // 6 * 32 * 8 = 1536 blocks
  const int ki = id >> 8;
  const int r  = id & 255;
  const int b  = r >> 3;
  const int mt = r & 7;
  const int k    = KSarr[ki];
  const int Kdim = F2T * k;
  const int Kp   = KpArr[ki];
  const int Mmax = S_ - k;                       // valid output rows 0..Mmax
  const int validRows = Mmax - mt * 16 + 1;      // rows beyond -> hardware-zeroed by TDM

#if __has_builtin(__builtin_amdgcn_tensor_load_to_lds)
  // ---- Tensor Data Mover: 2-D tile (16 rows x Kdim halfs, tensor row stride 42) -> LDS
  if (threadIdx.x < 32) {
    const unsigned long long ga =
        (unsigned long long)(size_t)(feat + (size_t)(b * S_ + mt * 16) * F2T);
    const unsigned ldsoff = (unsigned)(size_t)(void*)ldsA;  // low 32 bits = LDS offset
    v4u g0;
    g0[0] = 1u;                                             // count=1, user descriptor
    g0[1] = ldsoff;                                         // lds_addr
    g0[2] = (unsigned)(ga & 0xFFFFFFFFull);                 // global_addr[31:0]
    g0[3] = ((unsigned)(ga >> 32) & 0x01FFFFFFu) | 0x80000000u;  // addr[56:32] | type=2
    v8i g1;
    g1[0] = 1 << 16;                                        // data_size = 2 bytes
    g1[1] = (int)((unsigned)(Kdim & 0xFFFF) << 16);         // tensor_dim0[15:0] @bit48
    g1[2] = (int)((unsigned)(validRows & 0xFFFF) << 16);    // dim0 hi=0 | tensor_dim1[15:0]
    g1[3] = (int)((unsigned)(Kdim & 0xFFFF) << 16);         // dim1 hi=0 | tile_dim0
    g1[4] = 16;                                             // tile_dim1=16, tile_dim2=0
    g1[5] = F2T;                                            // tensor_dim0_stride = 42
    g1[6] = 0;
    g1[7] = 0;
    const v4i z4 = {0, 0, 0, 0};
#if defined(__clang_major__) && (__clang_major__ >= 23)
    const v8i z8 = {0, 0, 0, 0, 0, 0, 0, 0};
    __builtin_amdgcn_tensor_load_to_lds(g0, g1, z4, z4, z8, 0);
#else
    __builtin_amdgcn_tensor_load_to_lds(g0, g1, z4, z4, 0);
#endif
    __builtin_amdgcn_s_wait_tensorcnt(0);
  }
#else
  // fallback: cooperative staged copy with zero padding
  for (int t = threadIdx.x; t < 16 * Kdim; t += 128) {
    const int rr = t / Kdim, kkk = t % Kdim;
    const int mrow2 = mt * 16 + rr;
    ldsA[t] = (mrow2 <= Mmax) ? feat[((size_t)(b * S_ + mrow2)) * F2T + kkk] : (_Float16)0.f;
  }
#endif
  __syncthreads();

  const int lane = threadIdx.x & 31, wave = threadIdx.x >> 5;
  const int lo = lane & 15, hi = lane >> 4;
  const unsigned* A32 = (const unsigned*)ldsA;
  const _Float16* wT = convwT + woffT[ki];
  v8f acc[4] = {};
  for (int kk0 = 0; kk0 < Kp; kk0 += 32) {
    // A fragment: lane M=lo, chunks [kk0+8hi, +8) and [kk0+16+8hi, +8) from LDS
    AFrag af;
    const int base_dw = (lo * Kdim + kk0 + hi * 8) >> 1;
#pragma unroll
    for (int q2 = 0; q2 < 4; ++q2) {
      af.u[q2]     = A32[base_dw + q2];
      af.u[4 + q2] = A32[base_dw + 8 + q2];
    }
    if (kk0 + 32 > Kdim) {                       // tail-only predication (uniform branch)
#pragma unroll
      for (int e = 0; e < 16; ++e) {
        const int Ka = kk0 + (e < 8 ? e : e + 8) + hi * 8;
        if (Ka >= Kdim) af.h[e] = (_Float16)0.f;
      }
    }
#pragma unroll
    for (int q = 0; q < 4; ++q) {                // 4 N-tiles share one A fragment
      const int n = (wave * 4 + q) * 16 + lo;
      const _Float16* bp = wT + (size_t)n * Kp + kk0 + hi * 8;
      const v16h bf = __builtin_shufflevector(*(const v8h*)bp, *(const v8h*)(bp + 16), 0, 1, 2, 3, 4, 5, 6, 7, 8, 9, 10, 11, 12, 13, 14, 15);  // 16B aligned
      acc[q] = __builtin_amdgcn_wmma_f32_16x16x32_f16(false, af.v, false, bf,
                                                      (short)0, acc[q], false, false);
    }
  }
  const float* bias = cbias.p[ki];
#pragma unroll
  for (int q = 0; q < 4; ++q) {
    const int n = (wave * 4 + q) * 16 + lo;
    const float bv = bias[n];
#pragma unroll
    for (int rr = 0; rr < 8; ++rr) {
      const int row = mt * 16 + rr + (hi ? 8 : 0);
      if (row <= Mmax) {
        const float v = fmaxf(acc[q][rr] + bv, 0.f);  // ReLU; pooled pre-zeroed
        atomicMax((unsigned*)&pooled[(size_t)b * 1536 + ki * C_ + n], __float_as_uint(v));
      }
    }
  }
}

// ---- 5) pooled f32 -> f16 ----
__global__ void tohalf_kernel(const float* __restrict__ pooled, _Float16* __restrict__ z_h) {
  const int idx = blockIdx.x * blockDim.x + threadIdx.x;
  if (idx < B_ * 1536) z_h[idx] = (_Float16)pooled[idx];
}

// ---- 6) fc1: (32x1536)x(1536->512) WMMA, vector fragments ----
__global__ void fc1_wmma(const _Float16* __restrict__ z_h, const _Float16* __restrict__ w_h,
                         const float* __restrict__ bias, float* __restrict__ z2) {
  const int lane = threadIdx.x, lo = lane & 15, hi = lane >> 4;
  const int mt = blockIdx.x >> 5, nt = blockIdx.x & 31;
  const int row = mt * 16 + lo;
  const int n   = nt * 16 + lo;
  v8f acc = {};
  for (int kk0 = 0; kk0 < 1536; kk0 += 32) {
    union { v16h v; v8h c[2]; } af;
    af.c[0] = *(const v8h*)(z_h + (size_t)row * 1536 + kk0 + hi * 8);        // 16B aligned
    af.c[1] = *(const v8h*)(z_h + (size_t)row * 1536 + kk0 + 16 + hi * 8);
    const _Float16* bp = w_h + (size_t)n * 1536 + kk0 + hi * 8;
    const v16h bf = __builtin_shufflevector(*(const v8h*)bp, *(const v8h*)(bp + 16), 0, 1, 2, 3, 4, 5, 6, 7, 8, 9, 10, 11, 12, 13, 14, 15);  // 16B aligned
    acc = __builtin_amdgcn_wmma_f32_16x16x32_f16(false, af.v, false, bf,
                                                 (short)0, acc, false, false);
  }
  const float bv = bias[n];
#pragma unroll
  for (int rr = 0; rr < 8; ++rr) {
    const int m = mt * 16 + rr + (hi ? 8 : 0);
    z2[(size_t)m * NFC_ + n] = acc[rr] + bv;     // no ReLU between fc1/fc2 in reference
  }
}

// ---- 7) fc2: (32x512)x(512->5) scalar ----
__global__ void fc2_kernel(const float* __restrict__ z2, const float* __restrict__ w,
                           const float* __restrict__ bias, float* __restrict__ out) {
  const int idx = threadIdx.x;
  if (idx < B_ * NC_) {
    const int b = idx / NC_, c = idx % NC_;
    float s = bias[c];
    for (int h = 0; h < NFC_; ++h) s += z2[(size_t)b * NFC_ + h] * w[(size_t)c * NFC_ + h];
    out[idx] = s;
  }
}

extern "C" void kernel_launch(void* const* d_in, const int* in_sizes, int n_in,
                              void* d_out, int out_size, void* d_ws, size_t ws_size,
                              hipStream_t stream) {
  (void)in_sizes; (void)n_in; (void)out_size; (void)ws_size;
  const float* x    = (const float*)d_in[0];
  const float* w1   = (const float*)d_in[1];
  const float* b1   = (const float*)d_in[2];
  const float* w2   = (const float*)d_in[3];
  const float* b2   = (const float*)d_in[4];
  const float* fc1w = (const float*)d_in[5];
  const float* fc1b = (const float*)d_in[6];
  const float* fc2w = (const float*)d_in[7];
  const float* fc2b = (const float*)d_in[8];
  Ptr6 cw, cb;
  for (int i = 0; i < 6; ++i) {
    cw.p[i] = (const float*)d_in[9 + 2 * i];
    cb.p[i] = (const float*)d_in[10 + 2 * i];
  }
  char* ws = (char*)d_ws;
  float*    a_buf   = (float*)(ws + OFF_A);
  float*    b_buf   = (float*)(ws + OFF_B);
  _Float16* feat_h  = (_Float16*)(ws + OFF_FEATH);
  _Float16* convwT  = (_Float16*)(ws + OFF_CONVWT);
  _Float16* fc1w_h  = (_Float16*)(ws + OFF_FC1WH);
  float*    pooled  = (float*)(ws + OFF_POOL);
  _Float16* z_h     = (_Float16*)(ws + OFF_ZH);
  float*    z2      = (float*)(ws + OFF_Z2);

  proj_kernel<<<B_ * S_, 128, 0, stream>>>(x, w1, a_buf, b_buf);
  attn_kernel<<<B_ * S_, 128, 0, stream>>>(x, a_buf, b_buf, b1, w2, b2, feat_h);
  {
    const int total = 475136 + 786432 + B_ * 1536;   // convT + fc1 + pooled-zero
    prep_kernel<<<(total + 255) / 256, 256, 0, stream>>>(cw, fc1w, convwT, fc1w_h, pooled);
  }
  conv_wmma<<<6 * 32 * 8, 128, 0, stream>>>(feat_h, convwT, cb, pooled);
  tohalf_kernel<<<(B_ * 1536 + 255) / 256, 256, 0, stream>>>(pooled, z_h);
  fc1_wmma<<<2 * 32, 32, 0, stream>>>(z_h, fc1w_h, fc1b, z2);
  fc2_kernel<<<1, 256, 0, stream>>>(z2, fc2w, fc2b, (float*)d_out);
}
